// PointNetClassifier_57200374448389
// MI455X (gfx1250) — matrix operand order repaired and checked by reading, not compiled, not run on hardware
//
#include <hip/hip_runtime.h>
#include <hip/hip_bf16.h>
#include <math.h>

// ---------------------------------------------------------------------------
// PointNet GNN for MI455X (gfx1250, wave32, WMMA).
// Edge MLP runs on v_wmma_f32_16x16x32_f16; segment-max + ReLU fused into
// global_atomic_max_num_f32 against a zero-initialized aggregation buffer.
// ---------------------------------------------------------------------------

typedef _Float16 v16h __attribute__((ext_vector_type(16)));
typedef float    v8f  __attribute__((ext_vector_type(8)));

#define N_NODES 50000
#define N_EDGES 1600000
#define N_GRAPHS 64
#define HID 128
#define OUT_DIM 10
#define TILE_E 128         // edges per block
#define KP 160             // padded K stride for A/T tile in LDS (max 131 -> 160)

union Frag16 { uint4 u[2]; v16h v; };

__device__ __forceinline__ v8f splat8(float x) {
    v8f r;
#pragma unroll
    for (int i = 0; i < 8; ++i) r[i] = x;
    return r;
}

__device__ __forceinline__ void atomic_max_f32(float* p, float v) {
    // CDNA5 GLOBAL_ATOMIC_MAX_NUM_F32, no return (STOREcnt-tracked).
    asm volatile("global_atomic_max_num_f32 %0, %1, off"
                 :
                 : "v"((unsigned long long)(unsigned long long)(uintptr_t)p), "v"(v)
                 : "memory");
}

// ---------------------------------------------------------------------------
// Fill a float buffer with a constant (agg buffers must start at 0.0f so that
// atomic-max == relu(segment_max) with isolated nodes -> 0).
// ---------------------------------------------------------------------------
__global__ void fill_f32(float* __restrict__ p, int n, float v) {
    int i = blockIdx.x * blockDim.x + threadIdx.x;
    if (i < n) p[i] = v;
}

// ---------------------------------------------------------------------------
// Transpose + convert weights f32[K x N] -> f16[N x KPAD] (K-major rows, zero
// padded) so each WMMA B-fragment lane reads two contiguous 16-byte runs.
// ---------------------------------------------------------------------------
__global__ void conv_wT(const float* __restrict__ w, int K, int KPAD, int Ncols,
                        _Float16* __restrict__ out) {
    int idx = blockIdx.x * blockDim.x + threadIdx.x;
    if (idx >= Ncols * KPAD) return;
    int n = idx / KPAD;
    int k = idx - n * KPAD;
    out[idx] = (k < K) ? (_Float16)w[(size_t)k * Ncols + n] : (_Float16)0.0f;
}

// ---------------------------------------------------------------------------
// Edge-MLP + fused scatter-max.
//  - 256 threads = 8 waves, 128 edges per block (wave w owns edge rows 16w..).
//  - Phase 0: cooperative gather h[src] (f32->f16) + pos delta into LDS A.
//  - Phase 1: T = relu(A @ W1^T + b1) via WMMA, T overwrites wave's own rows.
//  - Phase 2: M = T @ W2^T + b2 via WMMA, atomic-max scatter into agg[dst].
// N_EDGES % TILE_E == 0, so EXEC is all-ones around every WMMA.
// ---------------------------------------------------------------------------
__global__ __launch_bounds__(256) void edge_mlp_wmma(
    const float* __restrict__ hin, int in_cols, int k1steps,
    const float* __restrict__ pos,
    const int* __restrict__ srcI, const int* __restrict__ dstI,
    const _Float16* __restrict__ W1t,  // [128][KP]  (K-major, padded)
    const float* __restrict__ b1,
    const _Float16* __restrict__ W2t,  // [128][128] (K-major)
    const float* __restrict__ b2,
    float* __restrict__ agg)           // [N_NODES][128], pre-zeroed
{
    __shared__ _Float16 Ash[TILE_E * KP];
    __shared__ int srcS[TILE_E];
    __shared__ int dstS[TILE_E];

    const int tid = threadIdx.x;
    const int e0  = blockIdx.x * TILE_E;

    if (tid < TILE_E) {
        srcS[tid] = srcI[e0 + tid];
        dstS[tid] = dstI[e0 + tid];
    }
    __syncthreads();

    // ---- Phase 0: gather features ------------------------------------------
    if (in_cols == HID) {
        // 128 edges x 32 float4 each
        for (int idx = tid; idx < TILE_E * 32; idx += 256) {
            int e = idx >> 5, j = idx & 31;
            const float4 v = ((const float4*)(hin + (size_t)srcS[e] * HID))[j];
            _Float16* d = &Ash[e * KP + j * 4];
            d[0] = (_Float16)v.x; d[1] = (_Float16)v.y;
            d[2] = (_Float16)v.z; d[3] = (_Float16)v.w;
        }
    } else {
        for (int idx = tid; idx < TILE_E * in_cols; idx += 256) {
            int e = idx / in_cols, j = idx - e * in_cols;
            Ash[e * KP + j] = (_Float16)hin[(size_t)srcS[e] * in_cols + j];
        }
    }
    if (tid < TILE_E) {
        int e = tid, s = srcS[e], d = dstS[e];
        Ash[e * KP + in_cols + 0] = (_Float16)(pos[3 * s + 0] - pos[3 * d + 0]);
        Ash[e * KP + in_cols + 1] = (_Float16)(pos[3 * s + 1] - pos[3 * d + 1]);
        Ash[e * KP + in_cols + 2] = (_Float16)(pos[3 * s + 2] - pos[3 * d + 2]);
        const int ktot = k1steps * 32;
        for (int k = in_cols + 3; k < ktot; ++k) Ash[e * KP + k] = (_Float16)0.0f;
    }
    __syncthreads();

    const int lane  = tid & 31;
    const int wv    = tid >> 5;
    const int lrow  = lane & 15;          // M row / N col within 16x16 tile
    const int kh    = lane >> 4;          // K half-select (A/B fragment layout)
    const int arow  = wv * 16 + lrow;     // this lane's A/T row in LDS

    // ---- Phase 1: T = relu(A @ W1^T + b1) ----------------------------------
    v8f acc[8];
#pragma unroll
    for (int n = 0; n < 8; ++n) acc[n] = splat8(b1[n * 16 + lrow]);

    for (int ks = 0; ks < k1steps; ++ks) {
        Frag16 af;
        const _Float16* ap = &Ash[arow * KP + ks * 32 + kh * 8];
        af.u[0] = *(const uint4*)(ap);
        af.u[1] = *(const uint4*)(ap + 16);
#pragma unroll
        for (int n = 0; n < 8; ++n) {
            Frag16 bf;
            const _Float16* bp = &W1t[(size_t)(n * 16 + lrow) * KP + ks * 32 + kh * 8];
            bf.u[0] = *(const uint4*)(bp);
            bf.u[1] = *(const uint4*)(bp + 16);
            acc[n] = __builtin_amdgcn_wmma_f32_16x16x32_f16(
                false, af.v, false, bf.v, (short)0, acc[n], false, false);
        }
    }
    // ReLU + f16, write T over this wave's own LDS rows (wave-private: no barrier)
#pragma unroll
    for (int n = 0; n < 8; ++n) {
#pragma unroll
        for (int r = 0; r < 8; ++r) {
            int trow = wv * 16 + r + kh * 8;
            Ash[trow * KP + n * 16 + lrow] = (_Float16)fmaxf(acc[n][r], 0.0f);
        }
    }

    // ---- Phase 2: M = T @ W2^T + b2 ----------------------------------------
    v8f acc2[8];
#pragma unroll
    for (int n = 0; n < 8; ++n) acc2[n] = splat8(b2[n * 16 + lrow]);

#pragma unroll
    for (int ks = 0; ks < 4; ++ks) {            // K = 128
        Frag16 af;
        const _Float16* ap = &Ash[arow * KP + ks * 32 + kh * 8];
        af.u[0] = *(const uint4*)(ap);
        af.u[1] = *(const uint4*)(ap + 16);
#pragma unroll
        for (int n = 0; n < 8; ++n) {
            Frag16 bf;
            const _Float16* bp = &W2t[(size_t)(n * 16 + lrow) * HID + ks * 32 + kh * 8];
            bf.u[0] = *(const uint4*)(bp);
            bf.u[1] = *(const uint4*)(bp + 16);
            acc2[n] = __builtin_amdgcn_wmma_f32_16x16x32_f16(
                false, af.v, false, bf.v, (short)0, acc2[n], false, false);
        }
    }

    // ---- Phase 3: fused segment-max scatter (relu folded into 0-init) ------
#pragma unroll
    for (int n = 0; n < 8; ++n) {
        const int col = n * 16 + lrow;
#pragma unroll
        for (int r = 0; r < 8; ++r) {
            int er   = wv * 16 + r + kh * 8;     // edge row (C/D layout M index)
            int node = dstS[er];
            atomic_max_f32(agg + (size_t)node * HID + col, acc2[n][r]);
        }
    }
}

// ---------------------------------------------------------------------------
// Global max-pool per graph: atomic-max h[node] into pooled[batch[node]].
// h >= 0 (post-ReLU) so 0-init matches the isfinite->0 semantics.
// ---------------------------------------------------------------------------
__global__ void pool_max(const float* __restrict__ h, const int* __restrict__ batch,
                         float* __restrict__ pooled) {
    int idx = blockIdx.x * blockDim.x + threadIdx.x;
    if (idx >= N_NODES * HID) return;
    int node = idx >> 7, col = idx & 127;
    atomic_max_f32(pooled + (size_t)batch[node] * HID + col, h[idx]);
}

// ---------------------------------------------------------------------------
// Classifier: logits = pooled @ cw + cb, softmax. 64x(128->10): VALU is fine.
// ---------------------------------------------------------------------------
__global__ void classifier(const float* __restrict__ pooled,
                           const float* __restrict__ cw, const float* __restrict__ cb,
                           float* __restrict__ out) {
    int g = threadIdx.x;
    if (g >= N_GRAPHS) return;
    float lg[OUT_DIM];
#pragma unroll
    for (int c = 0; c < OUT_DIM; ++c) {
        float s = cb[c];
        for (int k = 0; k < HID; ++k) s += pooled[g * HID + k] * cw[k * OUT_DIM + c];
        lg[c] = s;
    }
    float m = lg[0];
#pragma unroll
    for (int c = 1; c < OUT_DIM; ++c) m = fmaxf(m, lg[c]);
    float sum = 0.0f;
#pragma unroll
    for (int c = 0; c < OUT_DIM; ++c) { lg[c] = expf(lg[c] - m); sum += lg[c]; }
    float inv = 1.0f / sum;
#pragma unroll
    for (int c = 0; c < OUT_DIM; ++c) out[g * OUT_DIM + c] = lg[c] * inv;
}

// ---------------------------------------------------------------------------
extern "C" void kernel_launch(void* const* d_in, const int* in_sizes, int n_in,
                              void* d_out, int out_size, void* d_ws, size_t ws_size,
                              hipStream_t stream) {
    (void)in_sizes; (void)n_in; (void)out_size; (void)ws_size;

    const float* x    = (const float*)d_in[0];
    const float* pos  = (const float*)d_in[1];
    const int*   eidx = (const int*)d_in[2];
    const int*   bat  = (const int*)d_in[3];
    const float* w1[3] = { (const float*)d_in[4],  (const float*)d_in[8],  (const float*)d_in[12] };
    const float* b1[3] = { (const float*)d_in[5],  (const float*)d_in[9],  (const float*)d_in[13] };
    const float* w2[3] = { (const float*)d_in[6],  (const float*)d_in[10], (const float*)d_in[14] };
    const float* b2[3] = { (const float*)d_in[7],  (const float*)d_in[11], (const float*)d_in[15] };
    const float* cw = (const float*)d_in[16];
    const float* cb = (const float*)d_in[17];
    float* out = (float*)d_out;

    const int* srcI = eidx;
    const int* dstI = eidx + N_EDGES;

    // Workspace layout (fits in ~51.5 MB)
    const size_t NH = (size_t)N_NODES * HID;
    float*     hA     = (float*)d_ws;
    float*     hB     = hA + NH;
    _Float16*  W1t    = (_Float16*)(hB + NH);          // 3 x [128 x KP]
    _Float16*  W2t    = W1t + 3 * HID * KP;            // 3 x [128 x 128]
    float*     pooled = (float*)(W2t + 3 * HID * HID); // [64 x 128]

    const int K1[3]     = { 8 + 3, HID + 3, HID + 3 };  // 11, 131, 131
    const int KSTEPS[3] = { 1, 5, 5 };                  // padded-K / 32

    // Stage transposed f16 weights (one-time per launch, deterministic).
    for (int l = 0; l < 3; ++l) {
        conv_wT<<<(HID * KP + 255) / 256, 256, 0, stream>>>(
            w1[l], K1[l], KP, HID, W1t + (size_t)l * HID * KP);
        conv_wT<<<(HID * HID + 255) / 256, 256, 0, stream>>>(
            w2[l], HID, HID, HID, W2t + (size_t)l * HID * HID);
    }

    const int nodeElems = N_NODES * HID;
    const int fillGrid  = (nodeElems + 255) / 256;
    const int edgeGrid  = N_EDGES / TILE_E;             // 12500, exact

    // Layer 0: x -> hA
    fill_f32<<<fillGrid, 256, 0, stream>>>(hA, nodeElems, 0.0f);
    edge_mlp_wmma<<<edgeGrid, 256, 0, stream>>>(
        x, 8, KSTEPS[0], pos, srcI, dstI,
        W1t, b1[0], W2t, b2[0], hA);
    // Layer 1: hA -> hB
    fill_f32<<<fillGrid, 256, 0, stream>>>(hB, nodeElems, 0.0f);
    edge_mlp_wmma<<<edgeGrid, 256, 0, stream>>>(
        hA, HID, KSTEPS[1], pos, srcI, dstI,
        W1t + (size_t)1 * HID * KP, b1[1], W2t + (size_t)1 * HID * HID, b2[1], hB);
    // Layer 2: hB -> hA
    fill_f32<<<fillGrid, 256, 0, stream>>>(hA, nodeElems, 0.0f);
    edge_mlp_wmma<<<edgeGrid, 256, 0, stream>>>(
        hB, HID, KSTEPS[2], pos, srcI, dstI,
        W1t + (size_t)2 * HID * KP, b1[2], W2t + (size_t)2 * HID * HID, b2[2], hA);

    // Global max pool + classifier
    fill_f32<<<(N_GRAPHS * HID + 255) / 256, 256, 0, stream>>>(pooled, N_GRAPHS * HID, 0.0f);
    pool_max<<<(nodeElems + 255) / 256, 256, 0, stream>>>(hA, bat, pooled);
    classifier<<<1, 64, 0, stream>>>(pooled, cw, cb, out);
}
